// tf_encoder_54674933678705
// MI455X (gfx1250) — matrix operand order repaired and checked by reading, not compiled, not run on hardware
//
#include <hip/hip_runtime.h>
#include <hip/hip_bf16.h>
#include <math.h>

typedef __attribute__((ext_vector_type(16))) _Float16 v16h;
typedef __attribute__((ext_vector_type(8)))  _Float16 v8h;
typedef __attribute__((ext_vector_type(4)))  _Float16 v4h;
typedef __attribute__((ext_vector_type(8)))  float    v8f;
typedef __attribute__((ext_vector_type(4)))  float    v4f;

#define PI_F 3.14159265358979f
#define NO_RELU_CLIP (-3.0e38f)

// ---------------- model dims ----------------
#define BB   128
#define CC   9
#define TT   512
#define MM   64
#define DOB  40
#define DMODEL 360
#define DD   376      // D_MODEL + 16
#define NHEAD 4
#define DHH  94
#define DHP  96       // padded head dim
#define NHID 720

// =====================================================================
// WMMA fragment loaders (contiguous k order, same for A and B) — wide LDS loads
// =====================================================================
// A 16x32 f16: lane L: row=L&15, kb=(L>=16)?16:0; elements are one
// contiguous run K=kb..kb+15 -> one 32B LDS load.
__device__ inline v16h frag_a_lds(const _Float16* src, int ld) {
  int lane = threadIdx.x & 31;
  return *(const v16h*)(src + (size_t)(lane & 15) * ld + ((lane >> 4) << 4));
}

// B 32x16 f16 from a TRANSPOSED tile (src[row=col][k], contiguous in k):
// lane L: col=L&15, kb=(L>=16)?16:0, elements K=kb..kb+15 -> one 32B load.
__device__ inline v16h frag_b_lds_t(const _Float16* src, int ld) {
  int lane = threadIdx.x & 31;
  return *(const v16h*)(src + (size_t)(lane & 15) * ld + ((lane >> 4) << 4));
}

// =====================================================================
// WMMA GEMM: C[M,N] = max(A[M,K] @ B[K,N] + bias, clip), f32 io, f16 math
// block 256 threads = 8 waves, tile 128x128, BK=32; each wave: 32x64 (8 WMMA).
// REQUIRES: M % 128 == 0, K % 4 == 0, N % 4 == 0 (true for all call sites).
// bias must be a valid pointer (zero vector for "no bias").
// clip: 0.0f => fused ReLU, NO_RELU_CLIP => identity.
// Branch-free float4 staging; OOB handled by mask-multiply (keeps loads live).
// =====================================================================
__global__ __launch_bounds__(256) void gemm_wmma_kernel(
    const float* __restrict__ A, const float* __restrict__ Bm,
    const float* __restrict__ bias, float* __restrict__ C,
    int M, int N, int K, float clip) {
  __shared__ __attribute__((aligned(32))) _Float16 sA[128][32];   // [row][k]
  __shared__ __attribute__((aligned(32))) _Float16 sBt[128][32];  // [col][k]
  int tid  = threadIdx.x;
  int wave = tid >> 5;
  int wr   = wave & 3;        // 4 row groups of 32
  int wc   = wave >> 2;       // 2 col groups of 64
  int lane = tid & 31;
  int colL = lane & 15;
  int rofs = (lane >> 4) << 3;
  int bm = blockIdx.y * 128;
  int bn = blockIdx.x * 128;
  v8f acc[2][4];
#pragma unroll
  for (int i = 0; i < 2; ++i)
#pragma unroll
    for (int j = 0; j < 4; ++j) acc[i][j] = (v8f){0,0,0,0,0,0,0,0};

  for (int k0 = 0; k0 < K; k0 += 32) {
    // ---- A tile 128x32: 1024 float4 loads (rows always valid, M%128==0) ----
#pragma unroll
    for (int i = 0; i < 4; ++i) {
      int idx = tid + i * 256;          // 0..1023
      int r  = idx >> 3;                // 128 rows
      int c4 = (idx & 7) << 2;          // k offset 0,4,..,28
      int gk = k0 + c4;
      int gkc = min(gk, K - 4);         // aligned, always in-bounds
      const float* ap = A + (size_t)(bm + r) * K;
      v4f v = *(const v4f*)(ap + gkc);
      float msk = (gk < K) ? 1.0f : 0.0f;
      v4h hv;
      hv[0] = (_Float16)(v[0] * msk);
      hv[1] = (_Float16)(v[1] * msk);
      hv[2] = (_Float16)(v[2] * msk);
      hv[3] = (_Float16)(v[3] * msk);
      *(v4h*)&sA[r][c4] = hv;
      if (i == 0)  // prefetch next K-tile (clamped => always valid address)
        __builtin_prefetch((const void*)(ap + min(gk + 32, K - 4)), 0, 0);
    }
    // ---- B tile 32x128 -> transposed LDS: float4 along N, scatter stores ----
#pragma unroll
    for (int i = 0; i < 4; ++i) {
      int idx = tid + i * 256;
      int kk = idx >> 5;                // 32 k-rows (128/4 = 32 float4/row)
      int c4 = (idx & 31) << 2;         // col offset 0,4,..,124
      int gk = k0 + kk;
      int gc = bn + c4;
      int gkc = min(gk, K - 1);
      int gcc = min(gc, N - 4);
      const float* bpt = Bm + (size_t)gkc * N;
      v4f v = *(const v4f*)(bpt + gcc);
      float msk = (gk < K && gc < N) ? 1.0f : 0.0f;
      sBt[c4 + 0][kk] = (_Float16)(v[0] * msk);
      sBt[c4 + 1][kk] = (_Float16)(v[1] * msk);
      sBt[c4 + 2][kk] = (_Float16)(v[2] * msk);
      sBt[c4 + 3][kk] = (_Float16)(v[3] * msk);
      if (i == 0)
        __builtin_prefetch(
            (const void*)(Bm + (size_t)min(gk + 32, K - 1) * N + gcc), 0, 0);
    }
    __syncthreads();
    v16h fa0 = frag_a_lds(&sA[wr * 32][0], 32);
    v16h fa1 = frag_a_lds(&sA[wr * 32 + 16][0], 32);
#pragma unroll
    for (int j = 0; j < 4; ++j) {
      v16h fb = frag_b_lds_t(&sBt[wc * 64 + j * 16][0], 32);
      acc[0][j] = __builtin_amdgcn_wmma_f32_16x16x32_f16(
          false, fa0, false, fb, (short)0, acc[0][j], false, false);
      acc[1][j] = __builtin_amdgcn_wmma_f32_16x16x32_f16(
          false, fa1, false, fb, (short)0, acc[1][j], false, false);
    }
    __syncthreads();
  }
  // ---- epilogue: bias always valid, relu fused via clip ----
#pragma unroll
  for (int i = 0; i < 2; ++i)
#pragma unroll
    for (int j = 0; j < 4; ++j) {
      int gc = bn + wc * 64 + j * 16 + colL;
      if (gc < N) {
        float bv = bias[gc];
#pragma unroll
        for (int v = 0; v < 8; ++v) {
          int gr = bm + wr * 32 + i * 16 + rofs + v;   // always < M
          C[(size_t)gr * N + gc] = fmaxf(acc[i][j][v] + bv, clip);
        }
      }
    }
}

// =====================================================================
// Flash-style WMMA attention: one block (4 waves) per (b, head, 64-q tile)
// K tile row-major (contiguous along head-dim => wide B^T frags);
// V tile stored transposed (contiguous along key => wide B frags).
// =====================================================================
__global__ __launch_bounds__(128) void attn_wmma_kernel(
    const float* __restrict__ qkv,   // [B, T, 3*D]
    float* __restrict__ out) {       // [B, T, D]
  __shared__ __attribute__((aligned(32))) _Float16 sQ[64][DHP];
  __shared__ __attribute__((aligned(32))) _Float16 sK[64][DHP];
  __shared__ __attribute__((aligned(32))) _Float16 sVt[DHP][64];
  __shared__ __attribute__((aligned(32))) _Float16 sP[4][16][64];
  const float scale = 0.103141740f;  // 1/sqrt(94)
  int tid  = threadIdx.x;
  int wave = tid >> 5;
  int lane = tid & 31;
  int colL = lane & 15;
  int rofs = (lane >> 4) << 3;
  int blk = blockIdx.x;
  int qt = blk & 7;          // T/64 = 8 tiles
  int h  = (blk >> 3) & 3;
  int b  = blk >> 5;
  int q0 = qt * 64;
  size_t rstride = 3 * (size_t)DD;
  const float* qbp = qkv + (size_t)b * TT * rstride + (size_t)h * DHH;
  const float* kbp = qbp + DD;
  const float* vbp = qbp + 2 * DD;
  for (int idx = tid; idx < 64 * DHP; idx += 128) {
    int r = idx / DHP, d = idx % DHP;
    int dc = min(d, DHH - 1);
    float msk = (d < DHH) ? scale : 0.0f;
    float v = qbp[(size_t)(q0 + r) * rstride + dc];
    sQ[r][d] = (_Float16)(v * msk);
  }
  float m_run[8], l_run[8];
  v8f oacc[6];
#pragma unroll
  for (int v = 0; v < 8; ++v) { m_run[v] = -1e30f; l_run[v] = 0.0f; }
#pragma unroll
  for (int j = 0; j < 6; ++j) oacc[j] = (v8f){0,0,0,0,0,0,0,0};

  for (int kt = 0; kt < TT; kt += 64) {
    __syncthreads();   // prior P@V finished reading sVt; sQ visible on iter 0
    for (int idx = tid; idx < 64 * DHP; idx += 128) {
      int r = idx / DHP, d = idx % DHP;
      int dc = min(d, DHH - 1);
      float msk = (d < DHH) ? 1.0f : 0.0f;
      float kv = kbp[(size_t)(kt + r) * rstride + dc];
      float vv = vbp[(size_t)(kt + r) * rstride + dc];
      sK[r][d]  = (_Float16)(kv * msk);
      sVt[d][r] = (_Float16)(vv * msk);
    }
    __syncthreads();
    v8f sacc[4];
#pragma unroll
    for (int t = 0; t < 4; ++t) sacc[t] = (v8f){0,0,0,0,0,0,0,0};
#pragma unroll
    for (int dk = 0; dk < DHP; dk += 32) {
      v16h fa = frag_a_lds(&sQ[wave * 16][dk], DHP);
#pragma unroll
      for (int t = 0; t < 4; ++t) {
        v16h fb = frag_b_lds_t(&sK[t * 16][dk], DHP);  // K^T frag, contiguous
        sacc[t] = __builtin_amdgcn_wmma_f32_16x16x32_f16(
            false, fa, false, fb, (short)0, sacc[t], false, false);
      }
    }
    // online softmax over this 64-col block (row stats in half-wave)
#pragma unroll
    for (int v = 0; v < 8; ++v) {
      float tm = fmaxf(fmaxf(sacc[0][v], sacc[1][v]),
                       fmaxf(sacc[2][v], sacc[3][v]));
      tm = fmaxf(tm, __shfl_xor(tm, 1));
      tm = fmaxf(tm, __shfl_xor(tm, 2));
      tm = fmaxf(tm, __shfl_xor(tm, 4));
      tm = fmaxf(tm, __shfl_xor(tm, 8));
      float mn = fmaxf(m_run[v], tm);
      float alpha = __expf(m_run[v] - mn);
      float rs = 0.0f;
#pragma unroll
      for (int t = 0; t < 4; ++t) {
        float e = __expf(sacc[t][v] - mn);
        sacc[t][v] = e;
        rs += e;
      }
      rs += __shfl_xor(rs, 1);
      rs += __shfl_xor(rs, 2);
      rs += __shfl_xor(rs, 4);
      rs += __shfl_xor(rs, 8);
      l_run[v] = l_run[v] * alpha + rs;
      m_run[v] = mn;
#pragma unroll
      for (int j = 0; j < 6; ++j) oacc[j][v] *= alpha;
#pragma unroll
      for (int t = 0; t < 4; ++t)
        sP[wave][v + rofs][t * 16 + colL] = (_Float16)sacc[t][v];
    }
    __syncthreads();   // P visible (and S-phase done reading sK)
#pragma unroll
    for (int dk = 0; dk < 64; dk += 32) {
      v16h fp = frag_a_lds(&sP[wave][0][dk], 64);
#pragma unroll
      for (int j = 0; j < 6; ++j) {
        v16h fv = frag_b_lds_t(&sVt[j * 16][dk], 64);  // V frag, contiguous
        oacc[j] = __builtin_amdgcn_wmma_f32_16x16x32_f16(
            false, fp, false, fv, (short)0, oacc[j], false, false);
      }
    }
  }
#pragma unroll
  for (int j = 0; j < 6; ++j) {
    int c = j * 16 + colL;
    if (c < DHH) {
#pragma unroll
      for (int v = 0; v < 8; ++v) {
        int r = q0 + wave * 16 + rofs + v;
        float ov = oacc[j][v] / fmaxf(l_run[v], 1e-20f);
        out[((size_t)b * TT + r) * DD + h * DHH + c] = ov;
      }
    }
  }
}

// =====================================================================
// Spectral branch
// =====================================================================
__global__ __launch_bounds__(512) void dft_kernel(const float* __restrict__ x,
                                                  float* __restrict__ yfft) {
  __shared__ float xc[TT];
  int bc = blockIdx.x;                 // b*C + c
  int k = threadIdx.x;
  xc[k] = cosf(x[(size_t)bc * TT + k]);
  __syncthreads();
  float re = 0.0f, im = 0.0f;
  float w = -2.0f * PI_F * (float)k / (float)TT;
  for (int t = 0; t < TT; ++t) {
    float s, c;
    __sincosf(w * (float)t, &s, &c);
    re += xc[t] * c;
    im += xc[t] * s;
  }
  yfft[((size_t)bc * TT + k) * 2]     = re;
  yfft[((size_t)bc * TT + k) * 2 + 1] = im;
}

__global__ __launch_bounds__(64) void idft_chirp_kernel(
    const float* __restrict__ yfft, float* __restrict__ zfr) {
  __shared__ float wr[TT], wi[TT];
  int bc = blockIdx.x;
  int tid = threadIdx.x;
  for (int k = tid; k < TT; k += 64) {
    float yr = yfft[((size_t)bc * TT + k) * 2];
    float yi = yfft[((size_t)bc * TT + k) * 2 + 1];
    float ang = -PI_F * 0.5f * (float)k * (float)k / (float)TT; // FRAC=0.5
    float s, c;
    __sincosf(ang, &s, &c);
    wr[k] = yr * c - yi * s;
    wi[k] = yr * s + yi * c;
  }
  __syncthreads();
  int n = tid;                                    // first 64 outputs only
  float re = 0.0f, im = 0.0f;
  float w0 = 2.0f * PI_F * (float)n / (float)TT;
  for (int k = 0; k < TT; ++k) {
    float s, c;
    __sincosf(w0 * (float)k, &s, &c);
    re += wr[k] * c - wi[k] * s;
    im += wr[k] * s + wi[k] * c;
  }
  zfr[((size_t)bc * MM + n) * 2]     = re / (float)TT;
  zfr[((size_t)bc * MM + n) * 2 + 1] = im / (float)TT;
}

__global__ void modes_kernel(const float* __restrict__ zfr,
                             const float* __restrict__ w1r,
                             const float* __restrict__ w1i,
                             float* __restrict__ outft,   // complex interleaved
                             float* __restrict__ efpre) { // [B,C,2M]
  int idx = blockIdx.x * blockDim.x + threadIdx.x;
  if (idx >= BB * CC * MM) return;
  int m = idx & (MM - 1);
  int o = (idx >> 6) % CC;
  int b = idx / (MM * CC);
  float re = 0.0f, im = 0.0f;
  for (int i = 0; i < CC; ++i) {
    float zr = zfr[(((size_t)b * CC + i) * MM + m) * 2];
    float zi = zfr[(((size_t)b * CC + i) * MM + m) * 2 + 1];
    float wrr = w1r[((size_t)i * CC + o) * MM + m];
    float wii = w1i[((size_t)i * CC + o) * MM + m];
    re += zr * wrr - zi * wii;
    im += zr * wii + zi * wrr;
  }
  size_t ob = ((size_t)b * CC + o) * TT + m;
  outft[ob * 2]     = re;
  outft[ob * 2 + 1] = im;
  efpre[((size_t)b * CC + o) * (2 * MM) + m]      = sqrtf(re * re + im * im);
  efpre[((size_t)b * CC + o) * (2 * MM) + MM + m] = atan2f(im, re);
}

__global__ void conv_ef_kernel(const float* __restrict__ efpre,
                               const float* __restrict__ cw,
                               float* __restrict__ out) {  // [B,128]
  int idx = blockIdx.x * blockDim.x + threadIdx.x;
  if (idx >= BB * 2 * MM) return;
  int b = idx >> 7, j = idx & 127;
  float acc = 0.0f;
  for (int c = 0; c < CC; ++c)
    for (int dk = 0; dk < 3; ++dk) {
      int jj = j + dk - 1;
      if (jj >= 0 && jj < 2 * MM)
        acc += cw[c * 3 + dk] * efpre[((size_t)b * CC + c) * (2 * MM) + jj];
    }
  out[idx] = acc;
}

// BN over batch dim; optional ReLU after
__global__ void bn_feat_kernel(const float* __restrict__ in,
                               const float* __restrict__ g,
                               const float* __restrict__ b,
                               float* __restrict__ out,
                               int N, int F, int relu_after) {
  int f = blockIdx.x * blockDim.x + threadIdx.x;
  if (f >= F) return;
  float s = 0.0f, sq = 0.0f;
  for (int n = 0; n < N; ++n) {
    float v = in[(size_t)n * F + f];
    s += v; sq += v * v;
  }
  float m = s / (float)N;
  float var = sq / (float)N - m * m;
  float inv = rsqrtf(var + 1e-5f);
  for (int n = 0; n < N; ++n) {
    float v = (in[(size_t)n * F + f] - m) * inv * g[f] + b[f];
    if (relu_after) v = fmaxf(v, 0.0f);
    out[(size_t)n * F + f] = v;
  }
}

// =====================================================================
// Raindrop branch precompute
// =====================================================================
__global__ void hb_kernel(const float* __restrict__ x,
                          const float* __restrict__ Ru,
                          float* __restrict__ hb) {  // [(t*B+b)*40+d]
  int idx = blockIdx.x * blockDim.x + threadIdx.x;
  if (idx >= TT * BB * DOB) return;
  int d = idx % DOB;
  int b = (idx / DOB) % BB;
  int t = idx / (DOB * BB);
  float acc = 0.0f;
  for (int c = 0; c < CC; ++c) {
    float v = x[((size_t)b * CC + c) * TT + t] * Ru[c * DOB + d];
    acc += fmaxf(v, 0.0f);
  }
  hb[idx] = acc * (1.0f / (float)CC);
}

__global__ void build_z_kernel(const float* __restrict__ hg2,
                               float* __restrict__ z) {  // [B,T,376]
  int idx = blockIdx.x * blockDim.x + threadIdx.x;
  if (idx >= BB * TT * DD) return;
  int dch = idx % DD;
  int t = (idx / DD) % TT;
  int b = idx / (DD * TT);
  float v;
  if (dch < DMODEL) {
    v = hg2[((size_t)t * BB + b) * DOB + (dch % DOB)];
  } else {
    int i = dch - DMODEL;
    float ts = powf(100.0f, (float)(i & 7) / 7.0f);
    float st = (float)t / ts;
    v = (i < 8) ? sinf(st) : cosf(st);
  }
  z[idx] = v;
}

// residual add + LayerNorm over last dim
__global__ __launch_bounds__(128) void add_ln_kernel(
    float* __restrict__ z, const float* __restrict__ r,
    const float* __restrict__ g, const float* __restrict__ b, int D) {
  __shared__ float buf[DD];
  __shared__ float wsum[4], wsq[4];
  size_t row = blockIdx.x;
  int tid = threadIdx.x;
  float s = 0.0f, sq = 0.0f;
  for (int i = tid; i < D; i += 128) {
    float t = z[row * D + i] + r[row * D + i];
    buf[i] = t;
    s += t; sq += t * t;
  }
  for (int m = 16; m; m >>= 1) { s += __shfl_xor(s, m); sq += __shfl_xor(sq, m); }
  if ((tid & 31) == 0) { wsum[tid >> 5] = s; wsq[tid >> 5] = sq; }
  __syncthreads();
  float S  = wsum[0] + wsum[1] + wsum[2] + wsum[3];
  float SQ = wsq[0] + wsq[1] + wsq[2] + wsq[3];
  float mean = S / (float)D;
  float var  = SQ / (float)D - mean * mean;
  float inv  = rsqrtf(var + 1e-5f);
  for (int i = tid; i < D; i += 128)
    z[row * D + i] = (buf[i] - mean) * inv * g[i] + b[i];
}

__global__ void mean_t_kernel(const float* __restrict__ z,
                              float* __restrict__ out) { // [B,376]
  int idx = blockIdx.x * blockDim.x + threadIdx.x;
  if (idx >= BB * DD) return;
  int d = idx % DD, b = idx / DD;
  float s = 0.0f;
  for (int t = 0; t < TT; ++t) s += z[((size_t)b * TT + t) * DD + d];
  out[idx] = s * (1.0f / (float)TT);
}

__global__ __launch_bounds__(256) void concat_norm_kernel(
    const float* __restrict__ ef, const float* __restrict__ et,
    float* __restrict__ out) {
  __shared__ float f[640];
  __shared__ float part[8];
  int b = blockIdx.x, tid = threadIdx.x;
  float ss = 0.0f;
  for (int j = tid; j < 640; j += 256) {
    float v = (j < 128) ? ef[(size_t)b * 128 + j] : et[(size_t)b * 512 + (j - 128)];
    f[j] = v;
    ss += v * v;
  }
  for (int m = 16; m; m >>= 1) ss += __shfl_xor(ss, m);
  if ((tid & 31) == 0) part[tid >> 5] = ss;
  __syncthreads();
  float tot = 0.0f;
#pragma unroll
  for (int w = 0; w < 8; ++w) tot += part[w];
  float inv = 1.0f / fmaxf(sqrtf(tot), 1e-12f);
  for (int j = tid; j < 640; j += 256) out[(size_t)b * 640 + j] = f[j] * inv;
}

__global__ void zfill_kernel(float* __restrict__ p, size_t n) {
  size_t i = (size_t)blockIdx.x * blockDim.x + threadIdx.x;
  if (i < n) p[i] = 0.0f;
}

// =====================================================================
// Host-side orchestration
// =====================================================================
static inline void launch_gemm(const float* A, const float* B, const float* bias,
                               float* C, int M, int N, int K, float clip,
                               hipStream_t s) {
  dim3 grid((N + 127) / 128, M / 128);
  gemm_wmma_kernel<<<grid, 256, 0, s>>>(A, B, bias, C, M, N, K, clip);
}

extern "C" void kernel_launch(void* const* d_in, const int* in_sizes, int n_in,
                              void* d_out, int out_size, void* d_ws, size_t ws_size,
                              hipStream_t stream) {
  const float* x      = (const float*)d_in[0];
  const float* w1r    = (const float*)d_in[1];
  const float* w1i    = (const float*)d_in[2];
  const float* conv_w = (const float*)d_in[3];
  const float* bnf_g  = (const float*)d_in[4];
  const float* bnf_b  = (const float*)d_in[5];
  const float* R_u    = (const float*)d_in[6];
  const float* Wg1    = (const float*)d_in[7];
  const float* Wg2    = (const float*)d_in[8];
  const float* Wqkv   = (const float*)d_in[9];
  const float* bqkv   = (const float*)d_in[10];
  const float* Wo     = (const float*)d_in[11];
  const float* bo     = (const float*)d_in[12];
  const float* ln1g   = (const float*)d_in[13];
  const float* ln1b   = (const float*)d_in[14];
  const float* W1     = (const float*)d_in[15];
  const float* b1     = (const float*)d_in[16];
  const float* W2     = (const float*)d_in[17];
  const float* b2     = (const float*)d_in[18];
  const float* ln2g   = (const float*)d_in[19];
  const float* ln2b   = (const float*)d_in[20];
  const float* Wp     = (const float*)d_in[21];
  const float* bp     = (const float*)d_in[22];
  const float* bnp_g  = (const float*)d_in[23];
  const float* bnp_b  = (const float*)d_in[24];

  float* out = (float*)d_out;                  // [0,81920) fn ; then out_ft
  float* outft = out + (size_t)BB * 640;       // complex interleaved
  float* ws = (float*)d_ws;

  // workspace layout (floats)
  const size_t Z_SZ   = (size_t)BB * TT * DD;          // 24,641,536
  const size_t off_z    = 0;
  const size_t off_big  = off_z + Z_SZ;                // 73.9M region (qkv/ff1/early)
  const size_t off_attn = off_big + (size_t)BB * TT * 3 * DD;
  const size_t off_proj = off_attn + Z_SZ;
  const size_t off_ef   = off_proj + Z_SZ;             // [B,128]
  const size_t off_et   = off_ef + (size_t)BB * 128;   // [B,512]
  const size_t off_etp  = off_et + (size_t)BB * 512;   // [B,376]
  const size_t off_etr  = off_etp + (size_t)BB * DD;   // [B,512]
  const size_t off_zb   = off_etr + (size_t)BB * 512;  // zero bias [1152]
  // early-stage aliases inside off_big
  float* yfft  = ws + off_big;
  float* zfr   = ws + off_big + 1200000;
  float* efpre = ws + off_big + 1400000;
  float* efraw = ws + off_big + 1600000;
  float* hb    = ws + off_big + 1700000;
  float* hg1   = ws + off_big + 4400000;
  float* hg2   = ws + off_big + 7100000;

  float* zbuf  = ws + off_z;
  float* big   = ws + off_big;
  float* attn  = ws + off_attn;
  float* proj  = ws + off_proj;
  float* efb   = ws + off_ef;
  float* etb   = ws + off_et;
  float* etp   = ws + off_etp;
  float* etr   = ws + off_etr;
  float* zb    = ws + off_zb;

  // ---- zero fills: out_ft region + zero-bias vector ----
  {
    size_t n = (size_t)BB * CC * TT * 2;
    zfill_kernel<<<dim3((unsigned)((n + 255) / 256)), 256, 0, stream>>>(outft, n);
    zfill_kernel<<<dim3(5), 256, 0, stream>>>(zb, (size_t)1152);
  }

  // ---- spectral branch ----
  dft_kernel<<<BB * CC, TT, 0, stream>>>(x, yfft);
  idft_chirp_kernel<<<BB * CC, 64, 0, stream>>>(yfft, zfr);
  modes_kernel<<<(BB * CC * MM + 255) / 256, 256, 0, stream>>>(zfr, w1r, w1i,
                                                               outft, efpre);
  conv_ef_kernel<<<(BB * 128 + 255) / 256, 256, 0, stream>>>(efpre, conv_w, efraw);
  bn_feat_kernel<<<1, 128, 0, stream>>>(efraw, bnf_g, bnf_b, efb, BB, 128, 1);

  // ---- raindrop branch ----
  hb_kernel<<<(TT * BB * DOB + 255) / 256, 256, 0, stream>>>(x, R_u, hb);
  launch_gemm(hb, Wg1, zb, hg1, TT * BB, DOB, DOB, 0.0f, stream);
  launch_gemm(hg1, Wg2, zb, hg2, TT * BB, DOB, DOB, 0.0f, stream);
  build_z_kernel<<<(int)((Z_SZ + 255) / 256), 256, 0, stream>>>(hg2, zbuf);

  // ---- transformer layers ----
  const int MROWS = BB * TT;
  for (int l = 0; l < 3; ++l) {
    launch_gemm(zbuf, Wqkv + (size_t)l * DD * 3 * DD, bqkv + (size_t)l * 3 * DD,
                big, MROWS, 3 * DD, DD, NO_RELU_CLIP, stream);
    attn_wmma_kernel<<<BB * NHEAD * (TT / 64), 128, 0, stream>>>(big, attn);
    launch_gemm(attn, Wo + (size_t)l * DD * DD, bo + (size_t)l * DD,
                proj, MROWS, DD, DD, NO_RELU_CLIP, stream);
    add_ln_kernel<<<MROWS, 128, 0, stream>>>(zbuf, proj,
                                             ln1g + (size_t)l * DD,
                                             ln1b + (size_t)l * DD, DD);
    launch_gemm(zbuf, W1 + (size_t)l * DD * NHID, b1 + (size_t)l * NHID,
                big, MROWS, NHID, DD, 0.0f, stream);
    launch_gemm(big, W2 + (size_t)l * NHID * DD, b2 + (size_t)l * DD,
                proj, MROWS, DD, NHID, NO_RELU_CLIP, stream);
    add_ln_kernel<<<MROWS, 128, 0, stream>>>(zbuf, proj,
                                             ln2g + (size_t)l * DD,
                                             ln2b + (size_t)l * DD, DD);
  }

  // ---- pooling + projection + BN + concat/normalize ----
  mean_t_kernel<<<(BB * DD + 255) / 256, 256, 0, stream>>>(zbuf, etp);
  launch_gemm(etp, Wp, bp, etr, BB, TT, DD, 0.0f, stream);
  bn_feat_kernel<<<4, 128, 0, stream>>>(etr, bnp_g, bnp_b, etb, BB, TT, 0);
  concat_norm_kernel<<<BB, 256, 0, stream>>>(efb, etb, out);

  (void)in_sizes; (void)n_in; (void)out_size; (void)ws_size;
}
